// SimpleLSTM_39324720562493
// MI455X (gfx1250) — matrix operand order repaired and checked by reading, not compiled, not run on hardware
//
#include <hip/hip_runtime.h>
#include <hip/hip_bf16.h>

#define HID   2048
#define FOURH 8192
#define BATCH 128
#define VOC   95
#define VOCP  96          // padded to 6 WMMA N-tiles
#define TQ    162
#define TA    32

typedef __attribute__((ext_vector_type(16))) __bf16 v16bf;
typedef __attribute__((ext_vector_type(8)))  __bf16 v8bf;
typedef __attribute__((ext_vector_type(8)))  float  v8f;

// ---------------- precompute kernels ----------------

__global__ void cvt_bf16_kernel(const float* __restrict__ src, __bf16* __restrict__ dst, int n) {
    int i = blockIdx.x * blockDim.x + threadIdx.x;
    if (i < n) dst[i] = (__bf16)src[i];
}

__global__ void bias_sum_kernel(const float* __restrict__ a, const float* __restrict__ b,
                                float* __restrict__ o, int n) {
    int i = blockIdx.x * blockDim.x + threadIdx.x;
    if (i < n) o[i] = a[i] + b[i];
}

// ---------------- per-step GEMM: g = h @ Whh^T + bias + gx ----------------
// Whh_b : [4H][H] bf16 row-major (row n of Whh = column n of Whh^T -> contiguous in K)
// hb    : [B][H]  bf16 row-major (A-matrix), staged through double-buffered LDS
// mode  : 0 = gx none, 1 = gx token gather from Wih, 2 = gx dense buffer
__global__ __launch_bounds__(128)
void lstm_gemm_kernel(const __bf16* __restrict__ Whh_b,
                      const __bf16* __restrict__ hb,
                      const float*  __restrict__ bias,
                      const float*  __restrict__ Wih,
                      const int*    __restrict__ tok, int tokStride,
                      const float*  __restrict__ gxd,
                      int mode,
                      float* __restrict__ g_out) {
    // double-buffered LDS A-chunk: 128 rows x 32 K bf16, row stride padded to 40 (80B)
    __shared__ __bf16 As[2][BATCH * 40];
    __shared__ int    stok[BATCH];

    const int tid    = threadIdx.x;
    const int wave   = tid >> 5;        // 0..3
    const int lane   = tid & 31;
    const int laneLo = lane & 15;
    const int laneHi = lane >> 4;       // 0 or 1
    const int nTile  = blockIdx.x * 64 + wave * 16;   // 128 blocks cover N = 8192

    if (mode == 1) stok[tid] = tok[tid * tokStride];  // blockDim == 128 == BATCH

    // per-thread cooperative-copy coordinates (4 x 16B chunks per thread)
    int cRow[4], cQ[4];
    #pragma unroll
    for (int c2 = 0; c2 < 4; ++c2) {
        int idx  = tid + c2 * 128;
        cRow[c2] = idx >> 2;
        cQ[c2]   = idx & 3;
    }

    v8f acc[8];
    #pragma unroll
    for (int i = 0; i < 8; ++i) acc[i] = (v8f){0.f,0.f,0.f,0.f,0.f,0.f,0.f,0.f};

    // B operand: lane holds column N=nTile+laneLo, K = k0 + laneHi*16 .. +15 (contiguous)
    const __bf16* bsrc = Whh_b + (size_t)(nTile + laneLo) * HID + laneHi * 16;

#define PREFETCH(dstreg, K0N)                                                         \
    _Pragma("unroll")                                                                 \
    for (int c2 = 0; c2 < 4; ++c2)                                                    \
        dstreg[c2] = *(const v8bf*)(hb + (size_t)cRow[c2] * HID + (K0N) + cQ[c2] * 8);

#define STAGE(BUF, srcreg)                                                            \
    _Pragma("unroll")                                                                 \
    for (int c2 = 0; c2 < 4; ++c2)                                                    \
        *(v8bf*)(&As[BUF][cRow[c2] * 40 + cQ[c2] * 8]) = srcreg[c2];

#define COMPUTE(BUF, K0)                                                              \
    {                                                                                 \
        v16bf bfrag = *(const v16bf*)(bsrc + (K0));                                   \
        _Pragma("unroll")                                                             \
        for (int mt = 0; mt < 8; ++mt) {                                              \
            const __bf16* ap = &As[BUF][(mt * 16 + laneLo) * 40 + laneHi * 16];       \
            v8bf a0 = *(const v8bf*)(ap);                                             \
            v8bf a1 = *(const v8bf*)(ap + 8);                                         \
            v16bf afrag = __builtin_shufflevector(a0, a1,                             \
                0,1,2,3,4,5,6,7,8,9,10,11,12,13,14,15);                               \
            acc[mt] = __builtin_amdgcn_wmma_f32_16x16x32_bf16(                        \
                false, afrag, false, bfrag, (short)0, acc[mt], false, false);         \
        }                                                                             \
    }

    // prologue: stage chunk 0 into buffer 0
    {
        v8bf pre[4];
        PREFETCH(pre, 0)
        STAGE(0, pre)
    }
    __syncthreads();

    const int NK = HID / 32;   // 64 (even)
    for (int kk = 0; kk < NK; kk += 2) {
        // half 1: compute buf0 (chunk kk), prefetch chunk kk+1 -> buf1
        {
            v8bf nxt[4];
            PREFETCH(nxt, (kk + 1) * 32)
            COMPUTE(0, kk * 32)
            STAGE(1, nxt)
            __syncthreads();
        }
        // half 2: compute buf1 (chunk kk+1), prefetch chunk kk+2 -> buf0
        {
            v8bf nxt[4];
            const bool has = (kk + 2) < NK;
            if (has) { PREFETCH(nxt, (kk + 2) * 32) }
            COMPUTE(1, (kk + 1) * 32)
            if (has) { STAGE(0, nxt) }
            __syncthreads();
        }
    }
#undef PREFETCH
#undef STAGE
#undef COMPUTE

    // epilogue: C/D f32 layout -> VGPR r holds (M = r + laneHi*8, N = laneLo)
    const int   n  = nTile + laneLo;
    const float bn = bias[n];
    #pragma unroll
    for (int mt = 0; mt < 8; ++mt) {
        #pragma unroll
        for (int r = 0; r < 8; ++r) {
            int   m = mt * 16 + laneHi * 8 + r;
            float v = acc[mt][r] + bn;
            if (mode == 1)      v += Wih[(size_t)n * VOC + stok[m]]; // Wih^T[token]
            else if (mode == 2) v += gxd[(size_t)m * FOURH + n];
            g_out[(size_t)m * FOURH + n] = v;
        }
    }
}

// ---------------- elementwise LSTM gates ----------------
__global__ void lstm_point_kernel(const float* __restrict__ g,
                                  float* __restrict__ c,
                                  float* __restrict__ hf,
                                  __bf16* __restrict__ hb) {
    int idx = blockIdx.x * blockDim.x + threadIdx.x;   // b*HID + j
    if (idx >= BATCH * HID) return;
    int b = idx / HID, j = idx % HID;
    const float* gr = g + (size_t)b * FOURH;
    float gi = gr[j];
    float gf = gr[HID + j];
    float gg = gr[2 * HID + j];
    float go = gr[3 * HID + j];
    float si = 1.f / (1.f + expf(-gi));
    float sf = 1.f / (1.f + expf(-gf));
    float so = 1.f / (1.f + expf(-go));
    float cn = sf * c[idx] + si * tanhf(gg);
    float hn = so * tanhf(cn);
    c[idx]  = cn;
    hf[idx] = hn;
    hb[idx] = (__bf16)hn;
}

// ---------------- projection via WMMA: out[b][v] = h[b,:] . Wprj[v,:] ----------------
// Wp_b : [96][2048] bf16, row 95 zero-padded.  grid.x = 4 (M split), 192 threads (6 N-tiles)
__global__ __launch_bounds__(192)
void proj_wmma_kernel(const __bf16* __restrict__ hb,
                      const __bf16* __restrict__ Wp_b,
                      float* __restrict__ dst) {
    const int tid    = threadIdx.x;
    const int wave   = tid >> 5;        // 0..5
    const int lane   = tid & 31;
    const int laneLo = lane & 15;
    const int laneHi = lane >> 4;
    const int nTile  = wave * 16;
    const int mBase  = blockIdx.x * 32; // 4 blocks x 2 M-tiles

    v8f acc[2];
    acc[0] = (v8f){0.f,0.f,0.f,0.f,0.f,0.f,0.f,0.f};
    acc[1] = (v8f){0.f,0.f,0.f,0.f,0.f,0.f,0.f,0.f};

    const __bf16* bsrc = Wp_b + (size_t)(nTile + laneLo) * HID + laneHi * 16;

    for (int k0 = 0; k0 < HID; k0 += 32) {
        v16bf bfrag = *(const v16bf*)(bsrc + k0);
        #pragma unroll
        for (int mt = 0; mt < 2; ++mt) {
            const __bf16* ap = hb + (size_t)(mBase + mt * 16 + laneLo) * HID + k0 + laneHi * 16;
            v8bf a0 = *(const v8bf*)(ap);
            v8bf a1 = *(const v8bf*)(ap + 8);
            v16bf afrag = __builtin_shufflevector(a0, a1,
                0,1,2,3,4,5,6,7,8,9,10,11,12,13,14,15);
            acc[mt] = __builtin_amdgcn_wmma_f32_16x16x32_bf16(
                false, afrag, false, bfrag, (short)0, acc[mt], false, false);
        }
    }

    const int n = nTile + laneLo;
    if (n < VOC) {
        #pragma unroll
        for (int mt = 0; mt < 2; ++mt) {
            #pragma unroll
            for (int r = 0; r < 8; ++r) {
                int m = mBase + mt * 16 + laneHi * 8 + r;
                dst[(size_t)m * VOC + n] = acc[mt][r];
            }
        }
    }
}

// ---------------- gx = out0 @ Wih^T (K = 95, one-time, tiny) ----------------
__global__ void gx_dense_kernel(const float* __restrict__ out0,
                                const float* __restrict__ Wih,
                                float* __restrict__ gx) {
    int idx = blockIdx.x * blockDim.x + threadIdx.x;   // b*4H + n
    if (idx >= BATCH * FOURH) return;
    int b = idx / FOURH, n = idx % FOURH;
    const float* o = out0 + (size_t)b * VOC;
    const float* w = Wih + (size_t)n * VOC;
    float s = 0.f;
    #pragma unroll 5
    for (int v = 0; v < VOC; ++v) s += o[v] * w[v];
    gx[idx] = s;
}

// ---------------- host ----------------
extern "C" void kernel_launch(void* const* d_in, const int* in_sizes, int n_in,
                              void* d_out, int out_size, void* d_ws, size_t ws_size,
                              hipStream_t stream) {
    const int*   qs   = (const int*)d_in[0];   // [B][TQ]
    const int*   as_  = (const int*)d_in[2];   // [B][TA]
    const float* Wih  = (const float*)d_in[4]; // [4H][V]
    const float* Whh  = (const float*)d_in[5]; // [4H][H]
    const float* bih  = (const float*)d_in[6];
    const float* bhh  = (const float*)d_in[7];
    const float* Wprj = (const float*)d_in[8]; // [V][H]
    float* out = (float*)d_out;                // [31*B][V]

    char* ws = (char*)d_ws;
    size_t off = 0;
    auto alloc = [&](size_t bytes) -> void* {
        void* p = ws + off;
        off = (off + bytes + 255) & ~(size_t)255;
        return p;
    };
    __bf16* whh_b  = (__bf16*)alloc((size_t)FOURH * HID * 2);   // 32 MB (L2-resident)
    __bf16* wprj_b = (__bf16*)alloc((size_t)VOCP * HID * 2);    // padded projection weights
    __bf16* hb     = (__bf16*)alloc((size_t)BATCH * HID * 2);
    float*  hf     = (float*) alloc((size_t)BATCH * HID * 4);
    float*  c      = (float*) alloc((size_t)BATCH * HID * 4);
    float*  g      = (float*) alloc((size_t)BATCH * FOURH * 4);
    float*  gx     = (float*) alloc((size_t)BATCH * FOURH * 4);
    float*  bias   = (float*) alloc((size_t)FOURH * 4);

    // per-launch precompute — deterministic & idempotent
    cvt_bf16_kernel<<<(FOURH * HID + 255) / 256, 256, 0, stream>>>(Whh, whh_b, FOURH * HID);
    hipMemsetAsync(wprj_b, 0, (size_t)VOCP * HID * 2, stream);  // zero pad row 95
    cvt_bf16_kernel<<<(VOC * HID + 255) / 256, 256, 0, stream>>>(Wprj, wprj_b, VOC * HID);
    bias_sum_kernel<<<(FOURH + 255) / 256, 256, 0, stream>>>(bih, bhh, bias, FOURH);
    hipMemsetAsync(hb, 0, (size_t)BATCH * HID * 2, stream);
    hipMemsetAsync(hf, 0, (size_t)BATCH * HID * 4, stream);
    hipMemsetAsync(c,  0, (size_t)BATCH * HID * 4, stream);

    const dim3 gemmGrid(FOURH / 64), gemmBlk(128);
    const int  pwBlocks = (BATCH * HID) / 256;

    auto cell = [&](int mode, const int* tokp, int tokStride, const float* gxd) {
        lstm_gemm_kernel<<<gemmGrid, gemmBlk, 0, stream>>>(
            whh_b, hb, bias, Wih, tokp, tokStride, gxd, mode, g);
        lstm_point_kernel<<<pwBlocks, 256, 0, stream>>>(g, c, hf, hb);
    };

    // phase 1: 162 query steps, gx = Wih^T[qs[:, t]]
    for (int t = 0; t < TQ; ++t) cell(1, qs + t, TQ, nullptr);

    // phase 2: 15 free-running steps (gx = 0)
    for (int t = 0; t < 15; ++t) cell(0, nullptr, 0, nullptr);

    // out0 = h @ Wprj^T -> output slot 0
    proj_wmma_kernel<<<dim3(4), 192, 0, stream>>>(hb, wprj_b, out);

    // feedback step: gx = out0 @ Wih^T
    gx_dense_kernel<<<(BATCH * FOURH + 255) / 256, 256, 0, stream>>>(out, Wih, gx);
    cell(2, nullptr, 0, gx);

    // 30 answer steps: emit projection of h (pre-update), then cell with token as[:, j+2]
    for (int j = 0; j < 30; ++j) {
        proj_wmma_kernel<<<dim3(4), 192, 0, stream>>>(hb, wprj_b,
                                                      out + (size_t)(1 + j) * BATCH * VOC);
        cell(1, as_ + (j + 2), TA, nullptr);
    }
}
